// _AGRTMLPSwitchLiteTD_69664369541553
// MI455X (gfx1250) — matrix operand order repaired and checked by reading, not compiled, not run on hardware
//
#include <hip/hip_runtime.h>
#include <hip/hip_bf16.h>

// ---------------------------------------------------------------------------
// MI455X (gfx1250). Heavy GEMMs: v_wmma_f32_16x16x32_bf16, wave32, 32x32
// register block per wave (4 accumulators, A/B fragment reuse). f32 master
// activations kept for LN/statistics; LN affines folded into packed W1/b1.
// Head GEMMs sharing the same A are merged into one wide GEMM.
// ---------------------------------------------------------------------------

typedef __attribute__((ext_vector_type(16))) __bf16 v16bf;
typedef __attribute__((ext_vector_type(8)))  __bf16 v8bf;
typedef __attribute__((ext_vector_type(8)))  float  v8f;

#define B_ROWS   32768
#define D_NUM    512
#define N_CAT    8
#define EMB_D    32
#define STAT_STR 24   // per-row scalar scratch stride (f32)

__device__ __forceinline__ float wsum(float v){ for(int o=16;o;o>>=1) v += __shfl_xor(v,o,32); return v; }
__device__ __forceinline__ float wmax(float v){ for(int o=16;o;o>>=1) v  = fmaxf(v,__shfl_xor(v,o,32)); return v; }

// ---------------------------------------------------------------------------
// Weight packing: W (KxN f32 row-major) -> bf16 B-fragment layout.
// Tile (kt,nt) 32x16; per-lane 16 bf16 contiguous (two b128 loads).
// B lane map (ISA 7.12.2): n = lane&15, slot s -> k = kt*32 + (lane>>4)*8 + (s&7) + (s>>3)*16.
// Layout is nt-major, so merged GEMM segments pack via pointer offset.
// ---------------------------------------------------------------------------
__global__ void pack_w_kernel(const float* __restrict__ W, const float* __restrict__ lng,
                              int K, int N, int Kp, int Np, __bf16* __restrict__ out)
{
    size_t t = (size_t)blockIdx.x * 256 + threadIdx.x;
    size_t total = (size_t)Kp * Np;
    if (t >= total) return;
    int s    = (int)(t & 15);
    int lane = (int)((t >> 4) & 31);
    size_t tile = t >> 9;
    int nKT = Kp >> 5;
    int kt = (int)(tile % nKT);
    int nt = (int)(tile / nKT);
    int n = nt * 16 + (lane & 15);
    int k = kt * 32 + (lane >> 4) * 8 + (s & 7) + ((s >> 3) << 4);
    float v = 0.f;
    if (k < K && n < N) {
        v = W[(size_t)k * N + n];
        if (lng) v *= lng[k];
    }
    out[t] = (__bf16)v;
}

// b1'[n] = b1[n] + sum_k lnb[k]*W[k,n]  (zero-padded to Np)
__global__ void fold_bias_kernel(const float* __restrict__ b1, const float* __restrict__ lnb,
                                 const float* __restrict__ W, int K, int N, int Np,
                                 float* __restrict__ out)
{
    int n = blockIdx.x * 256 + threadIdx.x;
    if (n >= Np) return;
    float v = 0.f;
    if (n < N) {
        v = b1[n];
        if (lnb) for (int k = 0; k < K; ++k) v += lnb[k] * W[(size_t)k * N + n];
    }
    out[n] = v;
}

__global__ void cvt_bf16_kernel(const float* __restrict__ in, __bf16* __restrict__ out, size_t n)
{
    size_t t = (size_t)blockIdx.x * 256 + threadIdx.x;
    if (t < n) out[t] = (__bf16)in[t];
}

// emb gather -> feat cols [512, 768)
__global__ void emb_kernel(const int* __restrict__ xcat, const float* __restrict__ emb,
                           __bf16* __restrict__ feat)
{
    size_t t = (size_t)blockIdx.x * 256 + threadIdx.x;
    if (t >= (size_t)B_ROWS * 256) return;
    int row = (int)(t >> 8);
    int col = (int)(t & 255);
    int c = col >> 5, e = col & 31;
    int idx = xcat[(size_t)row * N_CAT + c];
    float v = emb[((size_t)c * 1000 + idx) * EMB_D + e];
    feat[(size_t)row * 768 + 512 + col] = (__bf16)v;
}

// ---------------------------------------------------------------------------
// WMMA GEMM, 32x32 per wave: 4 accumulators, 2 A-frags + 2 B-frags -> 4 WMMAs
// per k-step. mode 0: bf16 out. mode 1: f32 out. mode 2: gate epilogue
// (g=sigmoid -> outF ld512; feat[:,n]=bf16(xnum*g)).
// ---------------------------------------------------------------------------
__device__ __forceinline__ void gemm_store(float v, int m, int n, int act, int mode,
                                           __bf16* outBf, int ldBf, float* outF, int ldF,
                                           const float* xnum, __bf16* featOut)
{
    if (act == 1) v = fmaxf(v, 0.f);
    if (mode == 0) {
        outBf[(size_t)m * ldBf + n] = (__bf16)v;
    } else if (mode == 1) {
        outF[(size_t)m * ldF + n] = v;
    } else {
        float g = 1.f / (1.f + expf(-v));
        outF[(size_t)m * ldF + n] = g;
        featOut[(size_t)m * 768 + n] = (__bf16)(xnum[(size_t)m * D_NUM + n] * g);
    }
}

__global__ void wmma_gemm_kernel(const __bf16* __restrict__ A, const __bf16* __restrict__ Wp,
                                 const float* __restrict__ bias,
                                 int M, int Kp, int Np, int act, int mode,
                                 __bf16* __restrict__ outBf, int ldOutBf,
                                 float* __restrict__ outF, int ldOutF,
                                 const float* __restrict__ xnum, __bf16* __restrict__ featOut)
{
    int wave = blockIdx.x * (blockDim.x >> 5) + (threadIdx.x >> 5);
    int lane = threadIdx.x & 31;
    int nB = Np >> 5;                              // 32-wide blocks along N
    if (wave >= (M >> 5) * nB) return;             // wave-uniform: EXEC stays all-1 for WMMA
    int mb = wave / nB, nb = wave % nB;
    int hi = lane >> 4, l16 = lane & 15;
    int nKT = Kp >> 5;

    const __bf16* Arow0 = A + (size_t)(mb * 32 + l16) * Kp;      // A rows: M = lane&15 (+16)
    const __bf16* Arow1 = Arow0 + (size_t)16 * Kp;
    const __bf16* Bbase0 = Wp + ((size_t)(nb * 2)     * nKT) * 512;  // nt-major panels
    const __bf16* Bbase1 = Wp + ((size_t)(nb * 2 + 1) * nKT) * 512;
    __builtin_prefetch(Bbase0, 0, 2);              // warm L2 for the weight panel
    __builtin_prefetch(Bbase1, 0, 2);

    v8f acc00 = {}, acc01 = {}, acc10 = {}, acc11 = {};
    for (int kt = 0; kt < nKT; ++kt) {
        int k0 = kt * 32;
        v8bf x0 = *(const v8bf*)(Arow0 + k0 + 8 * hi);
        v8bf x1 = *(const v8bf*)(Arow0 + k0 + 16 + 8 * hi);
        v16bf a0 = __builtin_shufflevector(x0, x1, 0,1,2,3,4,5,6,7,8,9,10,11,12,13,14,15);
        v8bf y0 = *(const v8bf*)(Arow1 + k0 + 8 * hi);
        v8bf y1 = *(const v8bf*)(Arow1 + k0 + 16 + 8 * hi);
        v16bf a1 = __builtin_shufflevector(y0, y1, 0,1,2,3,4,5,6,7,8,9,10,11,12,13,14,15);
        const __bf16* bp0 = Bbase0 + ((size_t)kt * 32 + lane) * 16;
        v8bf p0 = *(const v8bf*)(bp0);
        v8bf p1 = *(const v8bf*)(bp0 + 8);
        v16bf b0 = __builtin_shufflevector(p0, p1, 0,1,2,3,4,5,6,7,8,9,10,11,12,13,14,15);
        const __bf16* bp1 = Bbase1 + ((size_t)kt * 32 + lane) * 16;
        v8bf q0 = *(const v8bf*)(bp1);
        v8bf q1 = *(const v8bf*)(bp1 + 8);
        v16bf b1 = __builtin_shufflevector(q0, q1, 0,1,2,3,4,5,6,7,8,9,10,11,12,13,14,15);
        acc00 = __builtin_amdgcn_wmma_f32_16x16x32_bf16(false, a0, false, b0, (short)0, acc00, false, false);
        acc01 = __builtin_amdgcn_wmma_f32_16x16x32_bf16(false, a0, false, b1, (short)0, acc01, false, false);
        acc10 = __builtin_amdgcn_wmma_f32_16x16x32_bf16(false, a1, false, b0, (short)0, acc10, false, false);
        acc11 = __builtin_amdgcn_wmma_f32_16x16x32_bf16(false, a1, false, b1, (short)0, acc11, false, false);
    }

    int n0 = nb * 32 + l16, n1 = n0 + 16;
    float bb0 = bias ? bias[n0] : 0.f;
    float bb1 = bias ? bias[n1] : 0.f;
#pragma unroll
    for (int r = 0; r < 8; ++r) {
        int m0 = mb * 32 + r + 8 * hi;             // C: VGPR r -> M = r + 8*hi
        int m1 = m0 + 16;
        gemm_store(acc00[r] + bb0, m0, n0, act, mode, outBf, ldOutBf, outF, ldOutF, xnum, featOut);
        gemm_store(acc01[r] + bb1, m0, n1, act, mode, outBf, ldOutBf, outF, ldOutF, xnum, featOut);
        gemm_store(acc10[r] + bb0, m1, n0, act, mode, outBf, ldOutBf, outF, ldOutF, xnum, featOut);
        gemm_store(acc11[r] + bb1, m1, n1, act, mode, outBf, ldOutBf, outF, ldOutF, xnum, featOut);
    }
}

// ---------------------------------------------------------------------------
// Row-wise matvec: stats[row, outOff..+dout) = H[row,:dh] @ W2 + b2. Wave/row.
// ---------------------------------------------------------------------------
__global__ void rowdot_kernel(const void* __restrict__ Hin, int isF32, int ldH, int dh,
                              const float* __restrict__ W2, const float* __restrict__ b2,
                              int dout, float* __restrict__ stats, int outOff, int M)
{
    int row  = blockIdx.x * (blockDim.x >> 5) + (threadIdx.x >> 5);
    int lane = threadIdx.x & 31;
    if (row >= M) return;
    float acc[4] = {0.f, 0.f, 0.f, 0.f};
    if (isF32) {
        const float* h = (const float*)Hin + (size_t)row * ldH;
        for (int j = lane; j < dh; j += 32) { float x = h[j];
            for (int d = 0; d < dout; ++d) acc[d] += x * W2[(size_t)j * dout + d]; }
    } else {
        const __bf16* h = (const __bf16*)Hin + (size_t)row * ldH;
        for (int j = lane; j < dh; j += 32) { float x = (float)h[j];
            for (int d = 0; d < dout; ++d) acc[d] += x * W2[(size_t)j * dout + d]; }
    }
    for (int d = 0; d < dout; ++d) acc[d] = wsum(acc[d]);
    if (lane == 0)
        for (int d = 0; d < dout; ++d)
            stats[(size_t)row * STAT_STR + outOff + d] = acc[d] + b2[d];
}

// ---------------------------------------------------------------------------
// LayerNorm of [H[row,:dbase] ++ extras] -> bf16, pad cols zeroed (affine folded
// into packed W1). extraSpec: 4 bits per extra = column in stats buffer.
// ---------------------------------------------------------------------------
__global__ void ln_kernel(const float* __restrict__ H, int ldH, int dbase,
                          const float* __restrict__ stats, int extraSpec, int nExtra,
                          __bf16* __restrict__ out, int ldOut, int M)
{
    int row  = blockIdx.x * (blockDim.x >> 5) + (threadIdx.x >> 5);
    int lane = threadIdx.x & 31;
    if (row >= M) return;
    const float* h = H + (size_t)row * ldH;
    float s = 0.f, s2 = 0.f;
    for (int j = lane; j < dbase; j += 32) { float x = h[j]; s += x; s2 += x * x; }
    if (lane < nExtra) {
        int col = (extraSpec >> (4 * lane)) & 15;
        float x = stats[(size_t)row * STAT_STR + col]; s += x; s2 += x * x;
    }
    s = wsum(s); s2 = wsum(s2);
    float dtot = (float)(dbase + nExtra);
    float mean = s / dtot;
    float var  = s2 / dtot - mean * mean;
    float inv  = rsqrtf(var + 1e-5f);
    __bf16* o = out + (size_t)row * ldOut;
    for (int j = lane; j < dbase; j += 32) o[j] = (__bf16)((h[j] - mean) * inv);
    for (int j = dbase + lane; j < ldOut; j += 32) {
        float v = 0.f;
        int e = j - dbase;
        if (e < nExtra) {
            int col = (extraSpec >> (4 * e)) & 15;
            v = (stats[(size_t)row * STAT_STR + col] - mean) * inv;
        }
        o[j] = (__bf16)v;
    }
}

// ---------------------------------------------------------------------------
// g statistics + top-k(8) + tiny cross network. Wave/row, shuffle reductions.
// stats: [1]=g_mean [2]=g_max [3]=g_std [4]=delta_cross
// ---------------------------------------------------------------------------
__global__ void stats_cross_kernel(const float* __restrict__ G, const float* __restrict__ xnum,
                                   float* __restrict__ stats,
                                   const float* Wsp, const float* bsp, const float* Wcp,
                                   const float* bcp, const float* lng, const float* lnb,
                                   const float* Wh1, const float* bh1, const float* Wh2,
                                   const float* bh2, const float* cross_scale, int M)
{
    int row  = blockIdx.x * (blockDim.x >> 5) + (threadIdx.x >> 5);
    int lane = threadIdx.x & 31;
    if (row >= M) return;
    const float* g = G + (size_t)row * D_NUM;

    float s = 0.f, s2 = 0.f, mx = -1e30f;
    for (int t = 0; t < 16; ++t) { float x = g[lane + 32 * t]; s += x; s2 += x * x; mx = fmaxf(mx, x); }
    s = wsum(s); s2 = wsum(s2); mx = wmax(mx);

    // top-8, lowest-index tiebreak
    unsigned used = 0;
    float topv[8]; int topi[8];
    for (int p = 0; p < 8; ++p) {
        float bv = -1e30f; int bi = 1 << 30;
        for (int t = 0; t < 16; ++t) {
            if (used & (1u << t)) continue;
            int j = lane + 32 * t; float x = g[j];
            if (x > bv || (x == bv && j < bi)) { bv = x; bi = j; }
        }
        for (int o = 16; o; o >>= 1) {
            float ov = __shfl_xor(bv, o, 32); int oi = __shfl_xor(bi, o, 32);
            if (ov > bv || (ov == bv && oi < bi)) { bv = ov; bi = oi; }
        }
        topv[p] = bv; topi[p] = bi;
        if ((bi & 31) == lane) used |= 1u << (bi >> 5);
    }

    if (lane == 0) {
        float mean = s / D_NUM;
        float var  = s2 / D_NUM - mean * mean;
        float* st = stats + (size_t)row * STAT_STR;
        st[1] = mean; st[2] = mx; st[3] = sqrtf(fmaxf(var, 0.f));

        float gs = 0.f;
        for (int p = 0; p < 8; ++p) gs += topv[p];
        float denom = gs + 1e-6f;
        float mw = 0.f;
        for (int p = 0; p < 8; ++p)
            mw += xnum[(size_t)row * D_NUM + topi[p]] * (topv[p] / denom);
        mw *= (1.f / 8.f);                          // pooled = mean over tokens

        float pooled[8], z[8];
        for (int d = 0; d < 8; ++d) pooled[d] = mw * Wsp[d] + bsp[d];
        for (int d = 0; d < 8; ++d) {
            float t = bcp[d];
            for (int e = 0; e < 8; ++e) t += pooled[e] * Wcp[e * 8 + d];
            z[d] = pooled[d] + pooled[d] * t;
        }
        float zm = 0.f, zv = 0.f;
        for (int d = 0; d < 8; ++d) zm += z[d];
        zm *= 0.125f;
        for (int d = 0; d < 8; ++d) { float dv = z[d] - zm; zv += dv * dv; }
        zv *= 0.125f;
        float zi = rsqrtf(zv + 1e-5f);
        float zn[8];
        for (int d = 0; d < 8; ++d) zn[d] = (z[d] - zm) * zi * lng[d] + lnb[d];
        float head = bh2[0];
        for (int hh = 0; hh < 16; ++hh) {
            float a = bh1[hh];
            for (int d = 0; d < 8; ++d) a += zn[d] * Wh1[d * 16 + hh];
            head += fmaxf(a, 0.f) * Wh2[hh];
        }
        st[4] = cross_scale[0] * tanhf(head);
    }
}

// alpha softmax (T=1.5), alpha stats, delta_expert. stats: [11]=amax [12]=ent [13]=gap [14]=dexp
__global__ void alpha_kernel(float* __restrict__ stats, const float* __restrict__ expert_scale, int M)
{
    int row = blockIdx.x * 256 + threadIdx.x;
    if (row >= M) return;
    float* st = stats + (size_t)row * STAT_STR;
    float r0 = st[8] / 1.5f, r1 = st[9] / 1.5f;
    float m = fmaxf(r0, r1);
    float e0 = expf(r0 - m), e1l = expf(r1 - m);
    float inv = 1.f / (e0 + e1l);
    float a0 = e0 * inv, a1 = e1l * inv;
    float e1 = st[6], e2 = st[7];
    st[11] = fmaxf(a0, a1);
    st[12] = -(a0 * logf(fmaxf(a0, 1e-8f)) + a1 * logf(fmaxf(a1, 1e-8f)));
    st[13] = fabsf(e1 - e2);
    st[14] = expert_scale[0] * tanhf(a0 * e1 + a1 * e2);
}

// final combine: mode softmax (T=1), beta sigmoid, output
__global__ void final_kernel(const float* __restrict__ stats, float* __restrict__ out, int M)
{
    int row = blockIdx.x * 256 + threadIdx.x;
    if (row >= M) return;
    const float* st = stats + (size_t)row * STAT_STR;
    float l0 = st[15], l1 = st[16], l2 = st[17], l3 = st[18];
    float m = fmaxf(fmaxf(l0, l1), fmaxf(l2, l3));
    float x0 = expf(l0 - m), x1 = expf(l1 - m), x2 = expf(l2 - m), x3 = expf(l3 - m);
    float inv = 1.f / (x0 + x1 + x2 + x3);
    float w_safe = x1 * inv, w_exp = x2 * inv, w_cross = x3 * inv;
    float beta = 1.f / (1.f + expf(-st[10]));
    out[row] = st[0] + beta * (w_safe * st[5] + w_exp * st[14] + w_cross * st[4]);
}

// ---------------------------------------------------------------------------
extern "C" void kernel_launch(void* const* d_in, const int* in_sizes, int n_in,
                              void* d_out, int out_size, void* d_ws, size_t ws_size,
                              hipStream_t stream)
{
    (void)in_sizes; (void)n_in; (void)out_size; (void)ws_size;
    const float* x_num = (const float*)d_in[0];
    const int*   x_cat = (const int*)d_in[1];
    // params flattened in setup_inputs() insertion order; rmlp = {ln_g,ln_b,W1,b1,W2,b2}
    const float *gW1 = (const float*)d_in[2],  *gb1 = (const float*)d_in[3];
    const float *gW2 = (const float*)d_in[4],  *gb2 = (const float*)d_in[5];
    const float *embp= (const float*)d_in[6];
    const float *bW1 = (const float*)d_in[7],  *bb1 = (const float*)d_in[8];
    const float *bW2 = (const float*)d_in[9],  *bb2 = (const float*)d_in[10];
    const float *baseW=(const float*)d_in[11], *baseb=(const float*)d_in[12];
    const float *sLg = (const float*)d_in[13], *sLb = (const float*)d_in[14];
    const float *sW1 = (const float*)d_in[15], *sb1 = (const float*)d_in[16];
    const float *sW2 = (const float*)d_in[17], *sb2 = (const float*)d_in[18];
    const float *e1Lg= (const float*)d_in[19], *e1Lb= (const float*)d_in[20];
    const float *e1W1= (const float*)d_in[21], *e1b1= (const float*)d_in[22];
    const float *e1W2= (const float*)d_in[23], *e1b2= (const float*)d_in[24];
    const float *e2Lg= (const float*)d_in[25], *e2Lb= (const float*)d_in[26];
    const float *e2W1= (const float*)d_in[27], *e2b1= (const float*)d_in[28];
    const float *e2W2= (const float*)d_in[29], *e2b2= (const float*)d_in[30];
    const float *rLg = (const float*)d_in[31], *rLb = (const float*)d_in[32];
    const float *rW1 = (const float*)d_in[33], *rb1 = (const float*)d_in[34];
    const float *rW2 = (const float*)d_in[35], *rb2 = (const float*)d_in[36];
    const float *btLg= (const float*)d_in[37], *btLb= (const float*)d_in[38];
    const float *btW1= (const float*)d_in[39], *btb1= (const float*)d_in[40];
    const float *btW2= (const float*)d_in[41], *btb2= (const float*)d_in[42];
    const float *mLg = (const float*)d_in[43], *mLb = (const float*)d_in[44];
    const float *mW1 = (const float*)d_in[45], *mb1 = (const float*)d_in[46];
    const float *mW2 = (const float*)d_in[47], *mb2 = (const float*)d_in[48];
    const float *cWsp= (const float*)d_in[49], *cbsp= (const float*)d_in[50];
    const float *cWcp= (const float*)d_in[51], *cbcp= (const float*)d_in[52];
    const float *cLg = (const float*)d_in[53], *cLb = (const float*)d_in[54];
    const float *cWh1= (const float*)d_in[55], *cbh1= (const float*)d_in[56];
    const float *cWh2= (const float*)d_in[57], *cbh2= (const float*)d_in[58];
    const float *exp_scale  = (const float*)d_in[59];
    const float *cross_scale= (const float*)d_in[60];

    // ---- workspace layout (256B aligned) ----
    char* ws = (char*)d_ws;
    size_t off = 0;
    auto alloc = [&](size_t bytes) -> char* {
        off = (off + 255) & ~(size_t)255;
        char* p = ws + off; off += bytes; return p;
    };
    const int M = B_ROWS;
    __bf16* XBF  = (__bf16*)alloc((size_t)M * 512  * 2);
    __bf16* FEAT = (__bf16*)alloc((size_t)M * 768  * 2);
    __bf16* GH1  = (__bf16*)alloc((size_t)M * 256  * 2);
    float*  Gf   = (float*) alloc((size_t)M * 512  * 4);
    __bf16* H1   = (__bf16*)alloc((size_t)M * 1024 * 2);
    float*  Hf   = (float*) alloc((size_t)M * 1024 * 4);
    __bf16* HN   = (__bf16*)alloc((size_t)M * 1024 * 2);
    __bf16* RN   = (__bf16*)alloc((size_t)M * 1056 * 2);
    __bf16* MN   = (__bf16*)alloc((size_t)M * 1056 * 2);
    __bf16* HDS  = (__bf16*)alloc((size_t)M * 1024 * 2);   // [safe 512 | e1 256 | e2 256]
    __bf16* RBH  = (__bf16*)alloc((size_t)M * 576  * 2);   // [router 288 | beta 288]
    __bf16* MH   = (__bf16*)alloc((size_t)M * 288  * 2);
    float*  STATS= (float*) alloc((size_t)M * STAT_STR * 4);
    __bf16* PW_G1  = (__bf16*)alloc((size_t)512 *256 *2);  float* PB_G1  = (float*)alloc(256*4);
    __bf16* PW_G2  = (__bf16*)alloc((size_t)256 *512 *2);  float* PB_G2  = (float*)alloc(512*4);
    __bf16* PW_BB1 = (__bf16*)alloc((size_t)768 *1024*2);  float* PB_BB1 = (float*)alloc(1024*4);
    __bf16* PW_BB2 = (__bf16*)alloc((size_t)1024*1024*2);  float* PB_BB2 = (float*)alloc(1024*4);
    __bf16* PW_HD  = (__bf16*)alloc((size_t)1024*1024*2);  float* PB_HD  = (float*)alloc(1024*4);
    __bf16* PW_RB  = (__bf16*)alloc((size_t)1056*576 *2);  float* PB_RB  = (float*)alloc(576*4);
    __bf16* PW_M   = (__bf16*)alloc((size_t)1056*288 *2);  float* PB_M   = (float*)alloc(288*4);

    auto packW = [&](const float* W, const float* lng, int K, int N, int Kp, int Np, __bf16* out) {
        size_t tot = (size_t)Kp * Np;
        pack_w_kernel<<<(int)((tot + 255) / 256), 256, 0, stream>>>(W, lng, K, N, Kp, Np, out);
    };
    auto foldB = [&](const float* b, const float* lnb, const float* W, int K, int N, int Np, float* out) {
        fold_bias_kernel<<<(Np + 255) / 256, 256, 0, stream>>>(b, lnb, W, K, N, Np, out);
    };
    // segment pointer inside a merged packed buffer: nt-major tiles of 512 elems
    auto seg = [&](__bf16* base, int colOff, int Kp) -> __bf16* {
        return base + (size_t)(colOff >> 4) * (size_t)(Kp >> 5) * 512;
    };

    // ---- pack weights (LN affine folded for rmlps; merged heads share A) ----
    packW(gW1, nullptr, 512, 256, 512, 256, PW_G1);    foldB(gb1, nullptr, gW1, 512, 256, 256, PB_G1);
    packW(gW2, nullptr, 256, 512, 256, 512, PW_G2);    foldB(gb2, nullptr, gW2, 256, 512, 512, PB_G2);
    packW(bW1, nullptr, 768, 1024, 768, 1024, PW_BB1); foldB(bb1, nullptr, bW1, 768, 1024, 1024, PB_BB1);
    packW(bW2, nullptr, 1024,1024, 1024,1024, PW_BB2); foldB(bb2, nullptr, bW2, 1024,1024, 1024, PB_BB2);
    // merged HN heads: [safe | e1 | e2]
    packW(sW1,  sLg,  1024, 512, 1024, 512, seg(PW_HD,   0, 1024));
    packW(e1W1, e1Lg, 1024, 256, 1024, 256, seg(PW_HD, 512, 1024));
    packW(e2W1, e2Lg, 1024, 256, 1024, 256, seg(PW_HD, 768, 1024));
    foldB(sb1,  sLb,  sW1,  1024, 512, 512, PB_HD);
    foldB(e1b1, e1Lb, e1W1, 1024, 256, 256, PB_HD + 512);
    foldB(e2b1, e2Lb, e2W1, 1024, 256, 256, PB_HD + 768);
    // merged RN heads: [router | beta], each padded 257 -> 288
    packW(rW1,  rLg,  1028, 257, 1056, 288, seg(PW_RB,   0, 1056));
    packW(btW1, btLg, 1028, 257, 1056, 288, seg(PW_RB, 288, 1056));
    foldB(rb1,  rLb,  rW1,  1028, 257, 288, PB_RB);
    foldB(btb1, btLb, btW1, 1028, 257, 288, PB_RB + 288);
    packW(mW1, mLg, 1031, 257, 1056, 288, PW_M);       foldB(mb1, mLb, mW1, 1031, 257, 288, PB_M);

    // ---- activations ----
    cvt_bf16_kernel<<<(int)(((size_t)M * 512 + 255) / 256), 256, 0, stream>>>(x_num, XBF, (size_t)M * 512);

    auto gemm = [&](const __bf16* A, const __bf16* Wp, const float* bias, int Kp, int Np,
                    int act, int mode, __bf16* oBf, int ldBf, float* oF, int ldF,
                    const float* xn, __bf16* feat) {
        int tiles = (M >> 5) * (Np >> 5);
        wmma_gemm_kernel<<<(tiles + 7) / 8, 256, 0, stream>>>(A, Wp, bias, M, Kp, Np, act, mode,
                                                              oBf, ldBf, oF, ldF, xn, feat);
    };
    const int WROW = M / 8;   // wave-per-row blocks (256 thr = 8 waves)
    const int TROW = M / 256; // thread-per-row blocks

    // gate MLP + sigmoid gating
    gemm(XBF, PW_G1, PB_G1, 512, 256, 1, 0, GH1, 256, nullptr, 0, nullptr, nullptr);
    gemm(GH1, PW_G2, PB_G2, 256, 512, 0, 2, nullptr, 0, Gf, 512, x_num, FEAT);
    // embeddings into feat
    emb_kernel<<<(int)(((size_t)M * 256 + 255) / 256), 256, 0, stream>>>(x_cat, embp, FEAT);
    // backbone
    gemm(FEAT, PW_BB1, PB_BB1, 768, 1024, 1, 0, H1, 1024, nullptr, 0, nullptr, nullptr);
    gemm(H1,   PW_BB2, PB_BB2, 1024,1024, 0, 1, nullptr, 0, Hf, 1024, nullptr, nullptr);
    // y_base
    rowdot_kernel<<<WROW, 256, 0, stream>>>(Hf, 1, 1024, 1024, baseW, baseb, 1, STATS, 0, M);
    // g stats + topk + cross
    stats_cross_kernel<<<WROW, 256, 0, stream>>>(Gf, x_num, STATS, cWsp, cbsp, cWcp, cbcp,
                                                 cLg, cLb, cWh1, cbh1, cWh2, cbh2, cross_scale, M);
    // normalized inputs (affine folded into W1): hn, router_in
    ln_kernel<<<WROW, 256, 0, stream>>>(Hf, 1024, 1024, STATS, 0,      0, HN, 1024, M);
    ln_kernel<<<WROW, 256, 0, stream>>>(Hf, 1024, 1024, STATS, 0x3210, 4, RN, 1056, M);
    // merged head GEMMs
    gemm(HN, PW_HD, PB_HD, 1024, 1024, 1, 0, HDS, 1024, nullptr, 0, nullptr, nullptr);
    gemm(RN, PW_RB, PB_RB, 1056, 576,  1, 0, RBH, 576,  nullptr, 0, nullptr, nullptr);
    rowdot_kernel<<<WROW, 256, 0, stream>>>(HDS,       0, 1024, 512, sW2,  sb2,  1, STATS, 5,  M);
    rowdot_kernel<<<WROW, 256, 0, stream>>>(HDS + 512, 0, 1024, 256, e1W2, e1b2, 1, STATS, 6,  M);
    rowdot_kernel<<<WROW, 256, 0, stream>>>(HDS + 768, 0, 1024, 256, e2W2, e2b2, 1, STATS, 7,  M);
    rowdot_kernel<<<WROW, 256, 0, stream>>>(RBH,       0, 576,  257, rW2,  rb2,  2, STATS, 8,  M);
    rowdot_kernel<<<WROW, 256, 0, stream>>>(RBH + 288, 0, 576,  257, btW2, btb2, 1, STATS, 10, M);
    // alpha / expert mixing scalars, then mode head
    alpha_kernel<<<TROW, 256, 0, stream>>>(STATS, exp_scale, M);
    ln_kernel<<<WROW, 256, 0, stream>>>(Hf, 1024, 1024, STATS, 0x0DCB3210, 7, MN, 1056, M);
    gemm(MN, PW_M, PB_M, 1056, 288, 1, 0, MH, 288, nullptr, 0, nullptr, nullptr);
    rowdot_kernel<<<WROW, 256, 0, stream>>>(MH, 0, 288, 257, mW2, mb2, 4, STATS, 15, M);
    // final combine
    final_kernel<<<TROW, 256, 0, stream>>>(STATS, (float*)d_out, M);
}
